// BasicGCN_1941325218147
// MI455X (gfx1250) — matrix operand order repaired and checked by reading, not compiled, not run on hardware
//
#include <hip/hip_runtime.h>

typedef __attribute__((ext_vector_type(16))) _Float16 v16h;
typedef __attribute__((ext_vector_type(8)))  _Float16 v8h;
typedef __attribute__((ext_vector_type(8)))  float    v8f;

#define LDS_K 136   // 128 + 8 halves pad (breaks 256B-stride bank aliasing)

// ---------------- small helpers ----------------
__device__ __forceinline__ float sigf(float x) { return 1.0f / (1.0f + __expf(-x)); }

__device__ __forceinline__ unsigned fenc(float f) {
    unsigned u = __float_as_uint(f);
    return (u & 0x80000000u) ? ~u : (u | 0x80000000u);
}
__device__ __forceinline__ float fdec(unsigned u) {
    unsigned v = (u & 0x80000000u) ? (u & 0x7FFFFFFFu) : ~u;
    return __uint_as_float(v);
}

// ---------------- WMMA fragment loaders (wave32, 16x16x32 f16) ----------------
// A 16x32 (MxK): lane<16 -> row=lane, halves 0..7 = K kb..kb+7, halves 8..15 = K kb+16..kb+23
//                lane>=16 -> row=lane-16, K base +8
__device__ __forceinline__ v16h load_fragA(const _Float16* lds, int row0, int kb) {
    const int lane = threadIdx.x & 31;
    const int r = row0 + (lane & 15);
    const int k = kb + ((lane >> 4) << 3);
    const _Float16* p = lds + r * LDS_K + k;
    v8h lo = *(const v8h*)(p);
    v8h hi = *(const v8h*)(p + 16);
    v16h f;
#pragma unroll
    for (int i = 0; i < 8; ++i) { f[i] = lo[i]; f[i + 8] = hi[i]; }
    return f;
}
// B 32x16 (KxN) from Bt stored [n][k]: lane<16 -> col=lane, halves 0..7 = K kb..kb+7, halves 8..15 = K kb+16..kb+23
//                                      lane>=16 -> col=lane-16, K base +8
__device__ __forceinline__ v16h load_fragB(const _Float16* lds, int col0, int kb) {
    const int lane = threadIdx.x & 31;
    const int c = col0 + (lane & 15);
    const int k = kb + ((lane >> 4) << 3);
    const _Float16* p = lds + c * LDS_K + k;
    v8h lo = *(const v8h*)(p);
    v8h hi = *(const v8h*)(p + 16);
    v16h f;
#pragma unroll
    for (int i = 0; i < 8; ++i) { f[i] = lo[i]; f[i + 8] = hi[i]; }
    return f;
}

// ---------------- one-time prep kernels ----------------
// Wt[i][j][k] = (f16) W[i][k][j]   (so B-frag K is contiguous)
__global__ void k_convw(const float* __restrict__ W, _Float16* __restrict__ Wt) {
    int idx = blockIdx.x * blockDim.x + threadIdx.x;
    if (idx >= 4 * 128 * 128) return;
    int i = idx >> 14, rem = idx & 16383, j = rem >> 7, k = rem & 127;
    Wt[idx] = (_Float16)W[(i << 14) + k * 128 + j];
}

__global__ void k_embed(const int* __restrict__ x, const float* __restrict__ emb,
                        float* __restrict__ h, int N) {
    int idx = blockIdx.x * blockDim.x + threadIdx.x;
    if (idx >= N * 32) return;
    int n = idx >> 5, c4 = (idx & 31) << 2;
    float4 v = *(const float4*)(emb + x[n] * 128 + c4);
    *(float4*)(h + (size_t)n * 128 + c4) = v;
}

__global__ void k_degree(const int* __restrict__ col, float* __restrict__ deg, int E) {
    int e = blockIdx.x * blockDim.x + threadIdx.x;
    if (e < E) atomicAdd(&deg[col[e]], 1.0f);
}

__global__ void k_dinv(float* __restrict__ dinv, int N) {
    int n = blockIdx.x * blockDim.x + threadIdx.x;
    if (n < N) dinv[n] = rsqrtf(dinv[n] + 1.0f);
}

__global__ void k_norm(const int* __restrict__ row, const int* __restrict__ col,
                       const float* __restrict__ dinv, float* __restrict__ nrm, int E) {
    int e = blockIdx.x * blockDim.x + threadIdx.x;
    if (e < E) nrm[e] = dinv[row[e]] * dinv[col[e]];
}

// ---------------- GEMM: xw[N,128] = h[N,128] @ W (via Wt f16), WMMA ----------------
__global__ __launch_bounds__(256) void k_gemm(const float* __restrict__ h,
                                              const _Float16* __restrict__ Wt,
                                              float* __restrict__ xw, int N) {
    __shared__ _Float16 Al[128 * LDS_K];
    __shared__ _Float16 Bl[128 * LDS_K];
    const int tid = threadIdx.x;
    const int base = blockIdx.x * 128;

    // stage A tile (f32 -> f16), zero-pad OOB rows
#pragma unroll
    for (int i = 0; i < 16; ++i) {
        int idx = tid + i * 256;            // 0..4095 float4-chunks
        int r = idx >> 5;
        int c4 = (idx & 31) << 2;
        float4 v = make_float4(0.f, 0.f, 0.f, 0.f);
        int g = base + r;
        if (g < N) v = *(const float4*)(h + (size_t)g * 128 + c4);
        _Float16* d = Al + r * LDS_K + c4;
        d[0] = (_Float16)v.x; d[1] = (_Float16)v.y;
        d[2] = (_Float16)v.z; d[3] = (_Float16)v.w;
    }
    // stage B (already f16, 128x128)
#pragma unroll
    for (int i = 0; i < 8; ++i) {
        int idx = tid + i * 256;            // 0..2047 8-half chunks
        int r = idx >> 4;
        int c8 = (idx & 15) << 3;
        *(v8h*)(Bl + r * LDS_K + c8) = *(const v8h*)(Wt + r * 128 + c8);
    }
    __syncthreads();

    const int wave = tid >> 5;
    const int row0 = wave * 16;
    v8f acc[8] = {};
    for (int kb = 0; kb < 128; kb += 32) {
        v16h a = load_fragA(Al, row0, kb);
#pragma unroll
        for (int j = 0; j < 8; ++j) {
            v16h b = load_fragB(Bl, j * 16, kb);
            acc[j] = __builtin_amdgcn_wmma_f32_16x16x32_f16(
                false, a, false, b, (short)0, acc[j], false, false);
        }
    }
    // C layout: VGPR v, lane l -> row = row0 + v + 8*(l>=16), col = j*16 + (l&15)
    const int lane = tid & 31;
    const int colb = lane & 15;
    const int rhalf = (lane >> 4) * 8;
#pragma unroll
    for (int j = 0; j < 8; ++j) {
#pragma unroll
        for (int v = 0; v < 8; ++v) {
            int g = base + row0 + rhalf + v;
            if (g < N) xw[(size_t)g * 128 + j * 16 + colb] = acc[j][v];
        }
    }
}

// ---------------- edge scatter: agg[col] += norm * xw[row] (wave per edge) ----------------
__global__ __launch_bounds__(256) void k_scatter(const int* __restrict__ row,
                                                 const int* __restrict__ col,
                                                 const float* __restrict__ nrm,
                                                 const float* __restrict__ xw,
                                                 float* __restrict__ agg, int E) {
    int e = (blockIdx.x * blockDim.x + threadIdx.x) >> 5;
    int lane = threadIdx.x & 31;
    if (e >= E) return;
    int r = row[e], c = col[e];
    float nv = nrm[e];
    float4 v = *(const float4*)(xw + (size_t)r * 128 + lane * 4);
    float* d = agg + (size_t)c * 128 + lane * 4;
    atomicAdd(d + 0, nv * v.x);
    atomicAdd(d + 1, nv * v.y);
    atomicAdd(d + 2, nv * v.z);
    atomicAdd(d + 3, nv * v.w);
}

// ---------------- h = relu(agg + self_norm*xw + b) ----------------
__global__ void k_relu(const float* __restrict__ agg, const float* __restrict__ xw,
                       const float* __restrict__ dinv, const float* __restrict__ bias,
                       float* __restrict__ h, int N) {
    int idx = blockIdx.x * blockDim.x + threadIdx.x;
    if (idx >= N * 32) return;
    int n = idx >> 5, c4 = (idx & 31) << 2;
    float di = dinv[n], sn = di * di;
    float4 a = *(const float4*)(agg + (size_t)n * 128 + c4);
    float4 w = *(const float4*)(xw + (size_t)n * 128 + c4);
    float4 b = *(const float4*)(bias + c4);
    float4 o;
    o.x = fmaxf(a.x + sn * w.x + b.x, 0.f);
    o.y = fmaxf(a.y + sn * w.y + b.y, 0.f);
    o.z = fmaxf(a.z + sn * w.z + b.z, 0.f);
    o.w = fmaxf(a.w + sn * w.w + b.w, 0.f);
    *(float4*)(h + (size_t)n * 128 + c4) = o;
}

// ---------------- Set2Set LSTM step (one graph per block, 512 gates) ----------------
__global__ __launch_bounds__(512) void k_lstm(const float* __restrict__ qstar,
                                              float* __restrict__ hh, float* __restrict__ cc,
                                              const float* __restrict__ Wih,
                                              const float* __restrict__ Whh,
                                              const float* __restrict__ bih,
                                              const float* __restrict__ bhh) {
    __shared__ float qs[256];
    __shared__ float hv[128];
    __shared__ float g[512];
    int b = blockIdx.x, tid = threadIdx.x;
    if (tid < 256) qs[tid] = qstar[(size_t)b * 256 + tid];
    if (tid < 128) hv[tid] = hh[(size_t)b * 128 + tid];
    __syncthreads();
    float acc = bih[tid] + bhh[tid];
    const float* wi = Wih + (size_t)tid * 256;
    const float* wh = Whh + (size_t)tid * 128;
#pragma unroll 4
    for (int k = 0; k < 256; ++k) acc += wi[k] * qs[k];
#pragma unroll 4
    for (int k = 0; k < 128; ++k) acc += wh[k] * hv[k];
    g[tid] = acc;
    __syncthreads();
    if (tid < 128) {
        float ig = sigf(g[tid]);
        float fg = sigf(g[128 + tid]);
        float gg = tanhf(g[256 + tid]);
        float og = sigf(g[384 + tid]);
        float c = fg * cc[(size_t)b * 128 + tid] + ig * gg;
        cc[(size_t)b * 128 + tid] = c;
        hh[(size_t)b * 128 + tid] = og * tanhf(c);
    }
}

// ---------------- attention: e[n] = <h[n], q[batch[n]]>, segment max ----------------
__global__ __launch_bounds__(256) void k_attn_e(const float* __restrict__ h,
                                                const float* __restrict__ q,
                                                const int* __restrict__ batch,
                                                float* __restrict__ ebuf,
                                                unsigned* __restrict__ m, int N) {
    int n = (blockIdx.x * blockDim.x + threadIdx.x) >> 5;
    int lane = threadIdx.x & 31;
    if (n >= N) return;
    int bi = batch[n];
    float4 hv = *(const float4*)(h + (size_t)n * 128 + lane * 4);
    float4 qv = *(const float4*)(q + (size_t)bi * 128 + lane * 4);
    float p = hv.x * qv.x + hv.y * qv.y + hv.z * qv.z + hv.w * qv.w;
#pragma unroll
    for (int o = 16; o > 0; o >>= 1) p += __shfl_xor(p, o, 32);
    if (lane == 0) {
        ebuf[n] = p;
        atomicMax(&m[bi], fenc(p));
    }
}

__global__ void k_attn_ex(const float* __restrict__ ebuf, const unsigned* __restrict__ m,
                          const int* __restrict__ batch, float* __restrict__ exb,
                          float* __restrict__ s, int N) {
    int n = blockIdx.x * blockDim.x + threadIdx.x;
    if (n >= N) return;
    int bi = batch[n];
    float x = __expf(ebuf[n] - fdec(m[bi]));
    exb[n] = x;
    atomicAdd(&s[bi], x);
}

__global__ __launch_bounds__(256) void k_attn_r(const float* __restrict__ h,
                                                const float* __restrict__ exb,
                                                const float* __restrict__ s,
                                                const int* __restrict__ batch,
                                                float* __restrict__ r, int N) {
    int n = (blockIdx.x * blockDim.x + threadIdx.x) >> 5;
    int lane = threadIdx.x & 31;
    if (n >= N) return;
    int bi = batch[n];
    float a = exb[n] / s[bi];
    float4 hv = *(const float4*)(h + (size_t)n * 128 + lane * 4);
    float* d = r + (size_t)bi * 128 + lane * 4;
    atomicAdd(d + 0, a * hv.x);
    atomicAdd(d + 1, a * hv.y);
    atomicAdd(d + 2, a * hv.z);
    atomicAdd(d + 3, a * hv.w);
}

__global__ void k_qstar(const float* __restrict__ hh, const float* __restrict__ r,
                        float* __restrict__ qstar) {
    int b = blockIdx.x, tid = threadIdx.x;
    float v = (tid < 128) ? hh[(size_t)b * 128 + tid] : r[(size_t)b * 128 + (tid - 128)];
    qstar[(size_t)b * 256 + tid] = v;
}

// ---------------- MLP head (one graph per block) ----------------
__global__ __launch_bounds__(128) void k_mlp(const float* __restrict__ qstar,
                                             const float* __restrict__ W0, const float* __restrict__ b0,
                                             const float* __restrict__ W1, const float* __restrict__ b1,
                                             const float* __restrict__ W3, const float* __restrict__ b3,
                                             float* __restrict__ out) {
    __shared__ float qs[256];
    __shared__ float z[128];
    __shared__ float z2[64];
    int b = blockIdx.x, tid = threadIdx.x;
    qs[tid] = qstar[(size_t)b * 256 + tid];
    qs[tid + 128] = qstar[(size_t)b * 256 + tid + 128];
    __syncthreads();
    {
        float acc = b0[tid];
        const float* w = W0 + (size_t)tid * 256;
#pragma unroll 4
        for (int k = 0; k < 256; ++k) acc += w[k] * qs[k];
        z[tid] = fmaxf(acc, 0.f);
    }
    __syncthreads();
    if (tid < 64) {
        float acc = b1[tid];
        const float* w = W1 + (size_t)tid * 128;
#pragma unroll 4
        for (int k = 0; k < 128; ++k) acc += w[k] * z[k];
        z2[tid] = fmaxf(acc, 0.f);
    }
    __syncthreads();
    if (tid == 0) {
        float acc = b3[0];
        for (int k = 0; k < 64; ++k) acc += W3[k] * z2[k];
        out[b] = acc;
    }
}

// ---------------- host ----------------
extern "C" void kernel_launch(void* const* d_in, const int* in_sizes, int n_in,
                              void* d_out, int out_size, void* d_ws, size_t ws_size,
                              hipStream_t stream) {
    const int* x      = (const int*)d_in[0];
    const int* eidx   = (const int*)d_in[1];
    const int* batch  = (const int*)d_in[2];
    const float* emb  = (const float*)d_in[3];
    const float* cW   = (const float*)d_in[4];
    const float* cb   = (const float*)d_in[5];
    const float* Wih  = (const float*)d_in[6];
    const float* Whh  = (const float*)d_in[7];
    const float* bih  = (const float*)d_in[8];
    const float* bhh  = (const float*)d_in[9];
    const float* W0   = (const float*)d_in[10];
    const float* b0   = (const float*)d_in[11];
    const float* W1   = (const float*)d_in[12];
    const float* b1   = (const float*)d_in[13];
    const float* W3   = (const float*)d_in[14];
    const float* b3   = (const float*)d_in[15];
    float* out        = (float*)d_out;

    const int N = in_sizes[0];
    const int E = in_sizes[1] / 2;
    const int B = out_size;
    const int* rowp = eidx;
    const int* colp = eidx + E;

    char* ws = (char*)d_ws;
    size_t off = 0;
    auto alloc = [&](size_t bytes) -> void* {
        void* p = ws + off;
        off = (off + bytes + 255) & ~(size_t)255;
        return p;
    };
    float*    h     = (float*)alloc((size_t)N * 128 * 4);
    float*    xw    = (float*)alloc((size_t)N * 128 * 4);
    float*    agg   = (float*)alloc((size_t)N * 128 * 4);
    float*    dinv  = (float*)alloc((size_t)N * 4);
    float*    nrm   = (float*)alloc((size_t)E * 4);
    _Float16* Wt    = (_Float16*)alloc((size_t)4 * 128 * 128 * 2);
    float*    qstar = (float*)alloc((size_t)B * 256 * 4);
    float*    hhb   = (float*)alloc((size_t)B * 128 * 4);
    float*    ccb   = (float*)alloc((size_t)B * 128 * 4);
    float*    ebuf  = (float*)alloc((size_t)N * 4);
    float*    exb   = (float*)alloc((size_t)N * 4);
    unsigned* mb    = (unsigned*)alloc((size_t)B * 4);
    float*    sb    = (float*)alloc((size_t)B * 4);
    float*    rb    = (float*)alloc((size_t)B * 128 * 4);

    // prep
    k_convw<<<(4 * 128 * 128 + 255) / 256, 256, 0, stream>>>(cW, Wt);
    k_embed<<<((size_t)N * 32 + 255) / 256, 256, 0, stream>>>(x, emb, h, N);
    hipMemsetAsync(dinv, 0, (size_t)N * 4, stream);
    k_degree<<<(E + 255) / 256, 256, 0, stream>>>(colp, dinv, E);
    k_dinv<<<(N + 255) / 256, 256, 0, stream>>>(dinv, N);
    k_norm<<<(E + 255) / 256, 256, 0, stream>>>(rowp, colp, dinv, nrm, E);

    // 4 GCN layers
    for (int i = 0; i < 4; ++i) {
        k_gemm<<<(N + 127) / 128, 256, 0, stream>>>(h, Wt + (size_t)i * 16384, xw, N);
        hipMemsetAsync(agg, 0, (size_t)N * 128 * 4, stream);
        k_scatter<<<((size_t)E * 32 + 255) / 256, 256, 0, stream>>>(rowp, colp, nrm, xw, agg, E);
        k_relu<<<((size_t)N * 32 + 255) / 256, 256, 0, stream>>>(agg, xw, dinv, cb + (size_t)i * 128, h, N);
    }

    // Set2Set (2 steps)
    hipMemsetAsync(qstar, 0, (size_t)B * 256 * 4, stream);
    hipMemsetAsync(hhb, 0, (size_t)B * 128 * 4, stream);
    hipMemsetAsync(ccb, 0, (size_t)B * 128 * 4, stream);
    for (int s = 0; s < 2; ++s) {
        k_lstm<<<B, 512, 0, stream>>>(qstar, hhb, ccb, Wih, Whh, bih, bhh);
        hipMemsetAsync(mb, 0, (size_t)B * 4, stream);
        hipMemsetAsync(sb, 0, (size_t)B * 4, stream);
        hipMemsetAsync(rb, 0, (size_t)B * 128 * 4, stream);
        k_attn_e<<<((size_t)N * 32 + 255) / 256, 256, 0, stream>>>(h, hhb, batch, ebuf, mb, N);
        k_attn_ex<<<(N + 255) / 256, 256, 0, stream>>>(ebuf, mb, batch, exb, sb, N);
        k_attn_r<<<((size_t)N * 32 + 255) / 256, 256, 0, stream>>>(h, exb, sb, batch, rb, N);
        k_qstar<<<B, 256, 0, stream>>>(hhb, rb, qstar);
    }

    // MLP head
    k_mlp<<<B, 128, 0, stream>>>(qstar, W0, b0, W1, b1, W3, b3, out);
}
